// SparseSelfAttentionConv2d_36876589204098
// MI455X (gfx1250) — matrix operand order repaired and checked by reading, not compiled, not run on hardware
//
#include <hip/hip_runtime.h>
#include <hip/hip_bf16.h>
#include <math.h>

// ---------------------------------------------------------------------------
// SparseSelfAttentionConv2d for MI455X (gfx1250, wave32, WMMA f32<-f16)
//
// B=32, CIN=512, N=256 (=16x16), COUT=2048, HEADS=8, D=256, SPARSITY=7
//
// Pipeline (all GEMMs via v_wmma_f32_16x16x32_f16, fp32 accumulate):
//   1) coeff_s  = top7-per-row(|coeff|) masked, sliced [:256,:256], as f16
//   2) q/k/v    = W{q,k,v}[2048x512] @ x_b[512x256] + bias       (f16 out)
//   3) attnRaw  = q^T k                    per (b,h)  [256x256]  (f16 out)
//   4) logits   = (attnRaw @ coeff_s) / scale[0]                 (f32 out)
//   5) P        = softmax(logits, axis=-1)                       (f16 out)
//   6) ctx[d,n] = sum_m v[d,m] * P[n,m]    per (b,h)             (f16 out)
//   7) out      = Wo[2048x2048] @ ctx_b + bo                     (f32 -> d_out)
// ---------------------------------------------------------------------------

typedef __attribute__((ext_vector_type(16))) _Float16 v16h;
typedef __attribute__((ext_vector_type(8)))  _Float16 v8h;
typedef __attribute__((ext_vector_type(4)))  _Float16 v4h;
typedef __attribute__((ext_vector_type(8)))  float    v8f;

union HFrag { v16h v; v8h h[2]; _Float16 e[16]; };
union FAcc  { v8f  v; float    e[8];  };

#define SPARSITY_ 7

// ---------------------------------------------------------------------------
// Top-7 per row of coeff[2048x2048] (|value|, ties -> lower index, matching
// jax.lax.top_k), keep entries whose column < 256, emit dense f16 [256x256].
// ---------------------------------------------------------------------------
__global__ __launch_bounds__(256)
void k_sparsify(const float* __restrict__ coeff, _Float16* __restrict__ coeffh)
{
    const int r = threadIdx.x;                 // 0..255
    float best[SPARSITY_];
    int   bidx[SPARSITY_];
#pragma unroll
    for (int t = 0; t < SPARSITY_; ++t) { best[t] = -1.0f; bidx[t] = -1; }

    const float* row = coeff + (long long)r * 2048;
    for (int c = 0; c < 2048; ++c) {
        float a = fabsf(row[c]);
        if (a > best[SPARSITY_ - 1]) {
            int j = SPARSITY_ - 1;
            while (j > 0 && a > best[j - 1]) {   // strict '>' keeps earlier index on ties
                best[j] = best[j - 1]; bidx[j] = bidx[j - 1]; --j;
            }
            best[j] = a; bidx[j] = c;
        }
    }

    _Float16* orow = coeffh + r * 256;
    for (int c = 0; c < 256; ++c) orow[c] = (_Float16)0.0f;
#pragma unroll
    for (int t = 0; t < SPARSITY_; ++t) {
        int c = bidx[t];
        if (c >= 0 && c < 256) orow[c] = (_Float16)row[c];
    }
}

// ---------------------------------------------------------------------------
// Generic strided GEMM:  C[bb][m][n] = sc * sum_k A(m,k)*B(k,n) + bias[m]
//   A element (m,k) at A[bb*saB + m*saR + k*saC]   (converted to f16 in LDS)
//   B element (k,n) at B[bb*sbB + k*sbR + n*sbC]
// Requirements: M%128==0, N%64==0, K%32==0 (all shapes here satisfy this).
// Block: 256 threads = 8 waves; block tile 128x64, K-step 32.
// Wave w computes rows [16w,16w+16) x 64 cols: one A fragment reused across
// four B fragments -> 4 v_wmma_f32_16x16x32_f16 per K-step per wave.
//
// LDS fragment layouts (per CDNA5 ISA 7.12.2, per-lane contiguous runs):
//   A-frag lane(m,half): e[0..7]=K 8h..8h+7, e[8..15]=K 16+8h..16+8h+7
//                        -> two ds_load_b128 from row-major As[m][k]
//   B-frag lane(n,half): e[0..7]=K 8h..8h+7, e[8..15]=K 16+8h..16+8h+7, fixed n
//                        -> two ds_load_b128 from TRANSPOSED Bt[n][k]
// ---------------------------------------------------------------------------
template <typename TA, typename TB, typename TOUT>
__global__ __launch_bounds__(256)
void k_gemm(int M, int N, int K,
            const TA* __restrict__ A, long long saR, long long saC, long long saB,
            const TB* __restrict__ Bm, long long sbR, long long sbC, long long sbB,
            const float* __restrict__ bias,
            const float* __restrict__ scaleDiv,
            TOUT* __restrict__ C, long long scB)
{
    __shared__ __align__(16) _Float16 As[128][40];  // row = 80B, 16B-aligned
    __shared__ __align__(16) _Float16 Bt[64][40];   // TRANSPOSED: Bt[n][k]

    const int tid  = threadIdx.x;
    const int lane = tid & 31;
    const int wv   = tid >> 5;          // 0..7
    const int half = (lane >> 4) & 1;
    const int l16  = lane & 15;
    const int wm   = wv * 16;

    const int       n0 = blockIdx.x * 64;
    const int       m0 = blockIdx.y * 128;
    const long long bb = blockIdx.z;

    const TA* Ab = A  + bb * saB;
    const TB* Bb = Bm + bb * sbB;

    FAcc acc[4];
#pragma unroll
    for (int g = 0; g < 4; ++g)
#pragma unroll
        for (int r = 0; r < 8; ++r) acc[g].e[r] = 0.0f;

    for (int k0 = 0; k0 < K; k0 += 32) {
        // ---- prefetch next A tile rows (lowers to global_prefetch_b8) ----
        if (k0 + 32 < K) {
            const TA* pf = Ab + (long long)(m0 + (tid >> 1)) * saR
                              + (long long)(k0 + 32 + 16 * (tid & 1)) * saC;
            __builtin_prefetch(pf, 0, 0);
        }

        // ---- stage A tile (128 x 32) into As[m][k] ----
        if (saC == 1) {
            if constexpr (sizeof(TA) == 4) {
                // f32 source, rows contiguous: float4 loads, packed f16 stores
                for (int base = tid * 4; base < 128 * 32; base += 256 * 4) {
                    int r = base >> 5, c = base & 31;
                    const float4 f =
                        *(const float4*)&Ab[(long long)(m0 + r) * saR + (k0 + c)];
                    v4h h; h.x = (_Float16)f.x; h.y = (_Float16)f.y;
                           h.z = (_Float16)f.z; h.w = (_Float16)f.w;
                    *(v4h*)&As[r][c] = h;
                }
            } else {
                // f16 source, rows contiguous: 16B loads + 16B LDS stores
                for (int base = tid * 8; base < 128 * 32; base += 256 * 8) {
                    int r = base >> 5, c = base & 31;
                    *(v8h*)&As[r][c] =
                        *(const v8h*)&Ab[(long long)(m0 + r) * saR + (k0 + c)];
                }
            }
        } else {
            for (int idx = tid; idx < 128 * 32; idx += 256) {
                int r = idx >> 5, c = idx & 31;
                As[r][c] = (_Float16)Ab[(long long)(m0 + r) * saR
                                        + (long long)(k0 + c) * saC];
            }
        }

        // ---- stage B tile (32k x 64n) transposed into Bt[n][k] ----
        if constexpr (sizeof(TB) == 2) {
            if (sbR == 1) {
                // k contiguous in global AND in Bt: 16B load + 16B LDS store
                // 64 n * 4 vectors of 8k = 256 vectors = 1 per thread
                int c = tid >> 2, rv = (tid & 3) * 8;
                *(v8h*)&Bt[c][rv] =
                    *(const v8h*)&Bb[(long long)(k0 + rv) + (long long)(n0 + c) * sbC];
            } else if (sbC == 1) {
                // n contiguous in global: 16B load + transposed b16 scatter
                int r = tid >> 3, cv = (tid & 7) * 8;
                v8h bvec = *(const v8h*)&Bb[(long long)(k0 + r) * sbR + (n0 + cv)];
#pragma unroll
                for (int i = 0; i < 8; ++i) Bt[cv + i][r] = bvec[i];
            } else {
                for (int idx = tid; idx < 32 * 64; idx += 256) {
                    int r = idx >> 6, c = idx & 63;
                    Bt[c][r] = (_Float16)Bb[(long long)(k0 + r) * sbR
                                            + (long long)(n0 + c) * sbC];
                }
            }
        } else {
            if (sbC == 1) {
                // f32, n contiguous: float4 load + transposed b16 scatter
                for (int base = tid * 4; base < 32 * 64; base += 256 * 4) {
                    int r = base >> 6, cv = base & 63;
                    const float4 f =
                        *(const float4*)&Bb[(long long)(k0 + r) * sbR + (n0 + cv)];
                    Bt[cv + 0][r] = (_Float16)f.x; Bt[cv + 1][r] = (_Float16)f.y;
                    Bt[cv + 2][r] = (_Float16)f.z; Bt[cv + 3][r] = (_Float16)f.w;
                }
            } else {
                for (int idx = tid; idx < 32 * 64; idx += 256) {
                    int r = idx >> 6, c = idx & 63;
                    Bt[c][r] = (_Float16)Bb[(long long)(k0 + r) * sbR
                                            + (long long)(n0 + c) * sbC];
                }
            }
        }
        __syncthreads();

        // ---- operand fragments: 10x ds_load_b128, 4x wmma ----
        HFrag a, b[4];
        a.h[0] = *(const v8h*)&As[wm + l16][8 * half];
        a.h[1] = *(const v8h*)&As[wm + l16][16 + 8 * half];
#pragma unroll
        for (int g = 0; g < 4; ++g) {
            b[g].h[0] = *(const v8h*)&Bt[g * 16 + l16][8 * half];
            b[g].h[1] = *(const v8h*)&Bt[g * 16 + l16][16 + 8 * half];
        }
#pragma unroll
        for (int g = 0; g < 4; ++g)
            acc[g].v = __builtin_amdgcn_wmma_f32_16x16x32_f16(
                false, a.v, false, b[g].v, (short)0, acc[g].v, false, false);
        __syncthreads();
    }

    const float sc = scaleDiv ? (1.0f / scaleDiv[0]) : 1.0f;
#pragma unroll
    for (int r = 0; r < 8; ++r) {
        int       ml   = r + 8 * half;            // C VGPR r -> M = r (+8 for upper half)
        long long grow = (long long)(m0 + wm + ml);
        float     bv   = bias ? bias[grow] : 0.0f;
        long long base = bb * scB + grow * (long long)N + n0;
#pragma unroll
        for (int g = 0; g < 4; ++g)
            C[base + g * 16 + l16] = (TOUT)(acc[g].e[r] * sc + bv);
    }
}

// ---------------------------------------------------------------------------
// Row softmax: in f32 [rows x 256] -> out f16 [rows x 256]; one wave per row.
// ---------------------------------------------------------------------------
__global__ __launch_bounds__(256)
void k_softmax(const float* __restrict__ in, _Float16* __restrict__ out, int nrows)
{
    const int  wv   = threadIdx.x >> 5;
    const int  lane = threadIdx.x & 31;
    const long long row = (long long)blockIdx.x * 8 + wv;
    if (row >= nrows) return;

    const float* r = in + row * 256;
    float v[8];
#pragma unroll
    for (int i = 0; i < 8; ++i) v[i] = r[lane + 32 * i];

    float m = v[0];
#pragma unroll
    for (int i = 1; i < 8; ++i) m = fmaxf(m, v[i]);
    for (int off = 16; off > 0; off >>= 1) m = fmaxf(m, __shfl_xor(m, off, 32));

    float s = 0.0f;
#pragma unroll
    for (int i = 0; i < 8; ++i) { v[i] = expf(v[i] - m); s += v[i]; }
    for (int off = 16; off > 0; off >>= 1) s += __shfl_xor(s, off, 32);
    const float inv = 1.0f / s;

    _Float16* o = out + row * 256;
#pragma unroll
    for (int i = 0; i < 8; ++i) o[lane + 32 * i] = (_Float16)(v[i] * inv);
}

// ---------------------------------------------------------------------------
// Launch
// ---------------------------------------------------------------------------
extern "C" void kernel_launch(void* const* d_in, const int* in_sizes, int n_in,
                              void* d_out, int out_size, void* d_ws, size_t ws_size,
                              hipStream_t stream)
{
    (void)in_sizes; (void)n_in; (void)out_size; (void)ws_size;

    const float* x     = (const float*)d_in[0];   // [32,512,16,16]
    const float* Wq    = (const float*)d_in[1];   // [2048,512]
    const float* bq    = (const float*)d_in[2];
    const float* Wk    = (const float*)d_in[3];
    const float* bk    = (const float*)d_in[4];
    const float* Wv    = (const float*)d_in[5];
    const float* bv    = (const float*)d_in[6];
    const float* Wo    = (const float*)d_in[7];   // [2048,2048]
    const float* bo    = (const float*)d_in[8];
    const float* coeff = (const float*)d_in[9];   // [2048,2048]
    const float* scale = (const float*)d_in[10];  // [1] = 16.0
    float* out = (float*)d_out;                   // [32,2048,16,16]

    // ---- workspace carve-up (bytes) ----
    char* ws = (char*)d_ws;
    _Float16* coeffh = (_Float16*)ws;                         // 128 KB
    _Float16* qh     = (_Float16*)(ws + (1ll << 20));         // 32 MB, [32][2048][256] f16
    _Float16* kh     = qh + 32ll * 2048 * 256;                // 32 MB
    _Float16* vh     = kh + 32ll * 2048 * 256;                // 32 MB
    _Float16* attn   = vh + 32ll * 2048 * 256;                // 32 MB: raw Q^T K, then P
    float*    logit  = (float*)(attn + 256ll * 256 * 256);    // 64 MB f32 logits
    _Float16* ctx    = (_Float16*)logit;                      // alias: f16 context (32 MB)

    const dim3 blk(256);
    const long long QKV_B = 2048ll * 256;   // per-batch q/k/v stride (elements)
    const long long BH_S  = 256ll * 256;    // per-(b,h) stride = 65536

    // 1) sparsify coeff -> f16 dense [256x256]
    k_sparsify<<<dim3(1), blk, 0, stream>>>(coeff, coeffh);

    // 2) projections: [2048x512]x[512x256] per batch, +bias, f16 out
    const dim3 gP(256 / 64, 2048 / 128, 32);
    k_gemm<float, float, _Float16><<<gP, blk, 0, stream>>>(2048, 256, 512,
        Wq, 512, 1, 0,  x, 256, 1, 512ll * 256, bq, nullptr, qh, QKV_B);
    k_gemm<float, float, _Float16><<<gP, blk, 0, stream>>>(2048, 256, 512,
        Wk, 512, 1, 0,  x, 256, 1, 512ll * 256, bk, nullptr, kh, QKV_B);
    k_gemm<float, float, _Float16><<<gP, blk, 0, stream>>>(2048, 256, 512,
        Wv, 512, 1, 0,  x, 256, 1, 512ll * 256, bv, nullptr, vh, QKV_B);

    // 3) attnRaw[n][m] = sum_d q[d][n] k[d][m]   (A = q^T via strides)
    const dim3 gA(256 / 64, 256 / 128, 256);
    k_gemm<_Float16, _Float16, _Float16><<<gA, blk, 0, stream>>>(256, 256, 256,
        qh, 1, 256, BH_S,  kh, 256, 1, BH_S, nullptr, nullptr, attn, BH_S);

    // 4) logits = (attnRaw @ coeff_s) / scale[0]   (f32 out for softmax)
    k_gemm<_Float16, _Float16, float><<<gA, blk, 0, stream>>>(256, 256, 256,
        attn, 256, 1, BH_S,  coeffh, 256, 1, 0, nullptr, scale, logit, BH_S);

    // 5) P = softmax(logits) -> f16 (overwrites attnRaw)
    k_softmax<<<dim3(65536 / 8), blk, 0, stream>>>(logit, attn, 65536);

    // 6) ctx[d][n] = sum_m v[d][m] * P[n][m]   (B = P^T via strides)
    k_gemm<_Float16, _Float16, _Float16><<<gA, blk, 0, stream>>>(256, 256, 256,
        vh, 256, 1, BH_S,  attn, 1, 256, BH_S, nullptr, nullptr, ctx, BH_S);

    // 7) out = Wo @ ctx + bo   (f32 -> d_out)
    k_gemm<float, _Float16, float><<<gP, blk, 0, stream>>>(2048, 256, 2048,
        Wo, 2048, 1, 0,  ctx, 256, 1, 2048ll * 256, bo, nullptr, out, 2048ll * 256);
}
